// STDPReservoir_30631706755665
// MI455X (gfx1250) — matrix operand order repaired and checked by reading, not compiled, not run on hardware
//
#include <hip/hip_runtime.h>
#include <hip/hip_bf16.h>
#include <stddef.h>

typedef __attribute__((ext_vector_type(16))) _Float16 v16h;
typedef __attribute__((ext_vector_type(8)))  float    v8f;

#define N_IN    128
#define N_RES   4096
#define T_STEPS 4096

// exp(-0.001/0.02) = exp(-0.05)
#define DECAY      0.95122942450071400910f
#define ONE_MDECAY 0.04877057549928599090f
// 0.12 / 4096
#define INH_SCALE  2.9296875e-5f

// CDNA5 async global->LDS path (ASYNCcnt), if this toolchain declares it.
// Probe result: parameter types are AS1 int* (global) and AS3 int* (LDS).
#if defined(__HIP_DEVICE_COMPILE__) && \
    __has_builtin(__builtin_amdgcn_global_load_async_to_lds_b32) && \
    __has_builtin(__builtin_amdgcn_s_wait_asynccnt)
#define HAVE_ASYNC_LDS 1
typedef __attribute__((address_space(1))) int gas_int;
typedef __attribute__((address_space(3))) int las_int;
#else
#define HAVE_ASYNC_LDS 0
#endif

// ---------------------------------------------------------------------------
// Repack pre_spikes (f32 0/1 -> f16, exact) into WMMA A-fragment order
// (16-bit A 16x32; contiguous k = 16*(lane/16) + h, same as pack_b). Packed so each lane's 16
// halves are contiguous (one 32B load).
// ---------------------------------------------------------------------------
__global__ void pack_a(const float* __restrict__ pre, _Float16* __restrict__ Apack) {
    int g    = blockIdx.x * blockDim.x + threadIdx.x;   // 0 .. 4096*128-1
    int h    = g & 15;
    int lane = (g >> 4) & 31;
    int kkmt = g >> 9;
    int kk   = kkmt & 3;          // k-chunk (K=32 each, 4 chunks = 128)
    int mt   = kkmt >> 2;         // m-tile (16 rows each, 256 tiles)
    int row  = mt * 16 + (lane & 15);
    int K    = kk * 32 + ((lane >> 4) << 4) + h;
    Apack[g] = (_Float16)pre[row * N_IN + K];
}

// B fragment order: 32x16 f16 per (n-tile, k-chunk): lanes 0..15: N=lane, K=h;
// lanes 16..31: N=lane-16, K=16+h.
__global__ void pack_b(const float* __restrict__ Win, _Float16* __restrict__ Bpack) {
    int g    = blockIdx.x * blockDim.x + threadIdx.x;   // 0 .. 128*4096-1
    int h    = g & 15;
    int lane = (g >> 4) & 31;
    int ntkk = g >> 9;
    int kk   = ntkk & 3;
    int nt   = ntkk >> 2;         // n-tile (16 cols each, 256 tiles)
    int col  = nt * 16 + (lane & 15);
    int K    = kk * 32 + ((lane >> 4) << 4) + h;
    Bpack[g] = (_Float16)Win[K * N_RES + col];
}

// ---------------------------------------------------------------------------
// Iin[4096,4096] = pre_spikes[4096,128] @ W_in[128,4096] via
// v_wmma_f32_16x16x32_f16: one 16x16 tile per wave, K=128 = 4 chained WMMAs.
// Staged into d_out (each step later consumes its row, then overwrites it).
// ---------------------------------------------------------------------------
__global__ void __launch_bounds__(256) gemm_iin(const _Float16* __restrict__ Apack,
                                                const _Float16* __restrict__ Bpack,
                                                float* __restrict__ out) {
    int mt   = blockIdx.x;               // 0..255
    int wave = threadIdx.x >> 5;         // 0..7
    int lane = threadIdx.x & 31;
    int nt   = blockIdx.y * 8 + wave;    // 0..255

    v8f c = {};
#pragma unroll
    for (int kk = 0; kk < 4; ++kk) {
        v16h a = *(const v16h*)(Apack + ((size_t)(mt * 4 + kk) * 32 + lane) * 16);
        v16h b = *(const v16h*)(Bpack + ((size_t)(nt * 4 + kk) * 32 + lane) * 16);
        c = __builtin_amdgcn_wmma_f32_16x16x32_f16(
                /*neg_a=*/false, a, /*neg_b=*/false, b,
                /*c_mod=*/(short)0, c, /*reuse_a=*/false, /*reuse_b=*/false);
    }
    // C/D layout: VGPR g -> M = g + (lane>=16 ? 8 : 0), N = lane&15
    int n     = nt * 16 + (lane & 15);
    int mbase = mt * 16 + ((lane >> 4) << 3);
#pragma unroll
    for (int g = 0; g < 8; ++g)
        out[(size_t)(mbase + g) * N_RES + n] = c[g];
}

// ---------------------------------------------------------------------------
// One full recurrence step in ONE kernel (last-block-finalizes pattern).
// Grid (4, 16): blockIdx.x -> 1024-col slice (float4/thread),
//               blockIdx.y -> 256-row chunk of W_rec.
// Stage 1 (all 64 blocks): spike-masked row-sum of L2-resident W_rec rows,
//   spike flags staged via GLOBAL_LOAD_ASYNC_TO_LDS (ASYNCcnt) when available.
// Stage 2 (the last block to arrive): LIF update for all neurons, deferred
//   reset + lateral inhibition from step t-1, deterministic LDS spike-count
//   reduction. pre/post traces are dead state (never affect output): dropped.
// ---------------------------------------------------------------------------
__global__ void __launch_bounds__(256) step_fused(const float* __restrict__ W,
                                                  float* __restrict__ spk,
                                                  float* __restrict__ part,
                                                  const float* __restrict__ vth,
                                                  float* __restrict__ vbuf,
                                                  float* __restrict__ out,
                                                  unsigned int* __restrict__ counter,
                                                  float* __restrict__ act_prev,
                                                  int t) {
    __shared__ float s_spk[256];
    __shared__ float s_red[256];
    __shared__ int   s_last;

    int tid     = threadIdx.x;
    int colbase = blockIdx.x * 1024 + tid * 4;
    int rowbase = blockIdx.y * 256;

#if HAVE_ASYNC_LDS
    __builtin_amdgcn_global_load_async_to_lds_b32(
        (gas_int*)(spk + rowbase + tid), (las_int*)&s_spk[tid], 0, 0);
    __builtin_amdgcn_s_wait_asynccnt(0);
#else
    s_spk[tid] = spk[rowbase + tid];
#endif
    __syncthreads();

    float4 acc = {0.f, 0.f, 0.f, 0.f};
    for (int i = 0; i < 256; ++i) {
        if (i + 1 < 256 && s_spk[i + 1] != 0.0f)
            __builtin_prefetch(W + (size_t)(rowbase + i + 1) * N_RES + colbase, 0, 1);
        if (s_spk[i] != 0.0f) {
            const float4 w = *(const float4*)(W + (size_t)(rowbase + i) * N_RES + colbase);
            acc.x += w.x; acc.y += w.y; acc.z += w.z; acc.w += w.w;
        }
    }
    *(float4*)(part + (size_t)blockIdx.y * N_RES + colbase) = acc;

    __threadfence();                       // flush partials to L2 (coherence point)
    __syncthreads();                       // all threads' stores precede arrival
    if (tid == 0) {
        unsigned int prev = atomicAdd(counter, 1u);
        s_last = (prev == gridDim.x * gridDim.y - 1u);
        if (s_last) *counter = 0u;         // everyone has arrived; reset for next step
    }
    __syncthreads();
    if (!s_last) return;

    // ---- fire stage: this block sees all partials (fence + arrival counter) ----
    float lact = (t > 0) ? *act_prev : 0.0f;
    float cnt  = 0.0f;
    for (int k = 0; k < 16; ++k) {
        int j = k * 256 + tid;
        float v  = vbuf[j];
        float sp = spk[j];                           // post(t-1)
        v = v * (1.0f - sp) - INH_SCALE * lact;      // deferred reset + inhibition

        float irec = 0.0f;
#pragma unroll
        for (int c = 0; c < 16; ++c)
            irec += part[(size_t)c * N_RES + j];

        float I = out[(size_t)t * N_RES + j] + irec; // Iin[t] staged in d_out
        v = DECAY * v + ONE_MDECAY * I;
        float post = (v >= vth[j]) ? 1.0f : 0.0f;
        cnt += post;
        vbuf[j] = v;                                 // pre-reset v (reset deferred)
        spk[j]  = post;
        out[(size_t)t * N_RES + j] = post;
    }
    // deterministic spike-count reduction (exact: integers in f32)
    s_red[tid] = cnt;
    __syncthreads();
#pragma unroll
    for (int s = 128; s > 0; s >>= 1) {
        if (tid < s) s_red[tid] += s_red[tid + s];
        __syncthreads();
    }
    if (tid == 0) *act_prev = s_red[0];
}

__global__ void init_state(float* __restrict__ spk, float* __restrict__ vbuf,
                           unsigned int* __restrict__ counter,
                           float* __restrict__ act_prev) {
    int i = blockIdx.x * blockDim.x + threadIdx.x;
    if (i < N_RES) { spk[i] = 0.0f; vbuf[i] = 0.0f; }
    if (i == 0)    { *counter = 0u; *act_prev = 0.0f; }
}

// ---------------------------------------------------------------------------
extern "C" void kernel_launch(void* const* d_in, const int* in_sizes, int n_in,
                              void* d_out, int out_size, void* d_ws, size_t ws_size,
                              hipStream_t stream) {
    const float* pre  = (const float*)d_in[0];   // [T, 128]
    const float* Win  = (const float*)d_in[1];   // [128, 4096]
    const float* Wrec = (const float*)d_in[2];   // [4096, 4096]
    const float* vth  = (const float*)d_in[3];   // [4096]
    float* out = (float*)d_out;                  // [T, 4096]

    char* ws = (char*)d_ws;
    _Float16*     Apack   = (_Float16*)ws;                    // 1 MB
    _Float16*     Bpack   = (_Float16*)(ws + (1u << 20));     // 1 MB
    float*        spk     = (float*)(ws + (2u << 20));        // 16 KB
    float*        vbuf    = spk + N_RES;                      // 16 KB
    float*        part    = vbuf + N_RES;                     // 16*4096*4 = 256 KB
    unsigned int* counter = (unsigned int*)(part + 16 * N_RES);
    float*        act     = (float*)(counter + 1);

    // Phase 0: fragment repack + state init
    pack_a<<<(N_RES * N_IN) / 256, 256, 0, stream>>>(pre, Apack);
    pack_b<<<(N_IN * N_RES) / 256, 256, 0, stream>>>(Win, Bpack);
    init_state<<<16, 256, 0, stream>>>(spk, vbuf, counter, act);

    // Phase 1: all input currents as one dense WMMA GEMM, staged in d_out
    gemm_iin<<<dim3(256, 32), 256, 0, stream>>>(Apack, Bpack, out);

    // Phase 2: sequential recurrence, ONE kernel per step
    for (int t = 0; t < T_STEPS; ++t)
        step_fused<<<dim3(4, 16), 256, 0, stream>>>(Wrec, spk, part, vth, vbuf,
                                                    out, counter, act, t);
}
